// GraphNN_32624571580791
// MI455X (gfx1250) — matrix operand order repaired and checked by reading, not compiled, not run on hardware
//
#include <hip/hip_runtime.h>

// ---------------------------------------------------------------------------
// GATv2 dual-stack GNN for MI455X (gfx1250).
// Dense linears via v_wmma_f32_16x16x32_bf16 (wave32), edge softmax/scatter
// via native f32 global atomics, segment-max via ordered-int atomicMax.
// ---------------------------------------------------------------------------

typedef __attribute__((ext_vector_type(16))) __bf16 v16bf;
typedef __attribute__((ext_vector_type(8)))  float  v8f;

static __device__ __forceinline__ unsigned short f2bf(float f) {
  unsigned int u = __float_as_uint(f);
  unsigned int r = u + 0x7FFFu + ((u >> 16) & 1u);   // round-to-nearest-even
  return (unsigned short)(r >> 16);
}
// monotone float -> int key for atomicMax-based segment max
static __device__ __forceinline__ int f2key(float f) {
  int i = __float_as_int(f);
  return i >= 0 ? i : (i ^ 0x7FFFFFFF);
}
static __device__ __forceinline__ float key2f(int k) {
  int i = k >= 0 ? k : (k ^ 0x7FFFFFFF);
  return __int_as_float(i);
}

// ------------------------------- utility kernels ---------------------------

__global__ void fill_u32_kernel(unsigned int* __restrict__ p, unsigned int v, size_t n) {
  size_t i = (size_t)blockIdx.x * blockDim.x + threadIdx.x;
  if (i < n) p[i] = v;
}

__global__ void cvt_f32_bf16_kernel(const float* __restrict__ in,
                                    unsigned short* __restrict__ out, size_t n) {
  size_t i = (size_t)blockIdx.x * blockDim.x + threadIdx.x;
  if (i < n) out[i] = f2bf(in[i]);
}

__global__ void add_bias_kernel(float* __restrict__ h, const float* __restrict__ bias, size_t n) {
  size_t i = (size_t)blockIdx.x * blockDim.x + threadIdx.x;
  if (i < n) h[i] += bias[i & 127];
}

// Pack a KxN(=128) fp32 weight matrix into the WMMA bf16 B-operand layout:
// Bp[kt][ntile][lane][j] with j<16, K_local = (j<8 ? (lane>>4)*8 + j : 16 + (lane>>4)*8 + j-8), n = ntile*16 + (lane&15)
__global__ void pack_b_kernel(const float* __restrict__ W, unsigned short* __restrict__ Bp, int K) {
  int t = blockIdx.x * blockDim.x + threadIdx.x;
  int total = (K >> 5) * 8 * 32;
  if (t >= total) return;
  int lane = t & 31;
  int nt   = (t >> 5) & 7;
  int kt   = t >> 8;
  int n    = nt * 16 + (lane & 15);
  int half = lane >> 4;
  unsigned short* o = Bp + (size_t)t * 16;
#pragma unroll
  for (int j = 0; j < 16; ++j) {
    int k = kt * 32 + (j < 8 ? half * 8 + j : 16 + half * 8 + (j - 8));
    o[j] = f2bf(W[(size_t)k * 128 + n]);
  }
}

// ------------------------------- WMMA GEMM ---------------------------------
// out[M x 128] = A[M x K](bf16) * Bpack + bias, with epilogues:
//   mode 0: out = acc + bias
//   mode 1: out = hres + elu(acc + bias)        (residual block)
//   mode 2: out = elu(acc + bias)               (fc1)
__global__ void __launch_bounds__(256) gemm_wmma_kernel(
    const unsigned short* __restrict__ Abf, const unsigned short* __restrict__ Bp,
    const float* __restrict__ bias, const float* __restrict__ hres,
    float* __restrict__ out, int M, int K, int mode) {
  int tile = blockIdx.x;          // 16-row strip
  int wave = threadIdx.x >> 5;    // column tile 0..7 (covers 128 cols)
  int lane = threadIdx.x & 31;
  int nloc = lane & 15;
  int half = lane >> 4;
  int row0 = tile * 16;
  int rowA = row0 + nloc; if (rowA > M - 1) rowA = M - 1;

  v8f acc = {0.f, 0.f, 0.f, 0.f, 0.f, 0.f, 0.f, 0.f};
  int ktiles = K >> 5;
  for (int kt = 0; kt < ktiles; ++kt) {
    union { v16bf v; uint4 u[2]; } A, B;
    // A 16x32 bf16: elems 0..7 -> K = half*8+j, elems 8..15 -> K = 16+half*8+j
    const unsigned short* ap = Abf + (size_t)rowA * K + kt * 32 + half * 8;
    A.u[0] = *(const uint4*)(ap);
    A.u[1] = *(const uint4*)(ap + 16);
    // B 32x16 bf16, prepacked: 16 contiguous bf16 per lane
    const unsigned short* bp = Bp + ((size_t)(kt * 8 + wave) * 32 + lane) * 16;
    B.u[0] = *(const uint4*)(bp);
    B.u[1] = *(const uint4*)(bp + 8);
    acc = __builtin_amdgcn_wmma_f32_16x16x32_bf16(false, A.v, false, B.v,
                                                  (short)0, acc, false, false);
  }

  int col = wave * 16 + nloc;
  float bb = bias[col];
#pragma unroll
  for (int r = 0; r < 8; ++r) {
    int m = row0 + half * 8 + r;
    if (m < M) {
      float v = acc[r] + bb;
      size_t o = (size_t)m * 128 + col;
      if (mode == 0) {
        out[o] = v;
      } else if (mode == 1) {
        float e = v > 0.f ? v : (__expf(v) - 1.f);
        out[o] = hres[o] + e;
      } else {
        out[o] = v > 0.f ? v : (__expf(v) - 1.f);
      }
    }
  }
}

// ------------------------------- edge kernels ------------------------------

// pass 1: logits[e,h] = att[h,:] . leaky(xl[src] + xr[dst] + eattr@lineW), seg-max
__global__ void edge_logits_kernel(
    const float* __restrict__ xl, const float* __restrict__ xr,
    const float* __restrict__ eattr, const float* __restrict__ lineW,
    const float* __restrict__ att, const int* __restrict__ src,
    const int* __restrict__ dst, float* __restrict__ logits,
    int* __restrict__ nmax, long long E, int ED) {
  long long t = (long long)blockIdx.x * blockDim.x + threadIdx.x;
  if (t >= E * 8) return;
  long long e = t >> 3;
  int h = (int)(t & 7);
  int s = src[e], d = dst[e];
  float ea[12];
  for (int k = 0; k < ED; ++k) ea[k] = eattr[e * ED + k];
  const float* pa  = att + h * 16;
  const float* pxl = xl + (size_t)s * 128 + h * 16;
  const float* pxr = xr + (size_t)d * 128 + h * 16;
  float logit = 0.f;
#pragma unroll
  for (int i = 0; i < 16; ++i) {
    int c = h * 16 + i;
    float em = 0.f;
    for (int k = 0; k < ED; ++k) em += ea[k] * lineW[k * 128 + c];
    float m = pxl[i] + pxr[i] + em;
    m = m > 0.f ? m : 0.2f * m;        // leaky_relu(0.2)
    logit += pa[i] * m;
  }
  logits[t] = logit;
  atomicMax(nmax + (size_t)d * 8 + h, f2key(logit));
}

// pass 2: ex = exp(logit - max[dst]); denom[dst] += ex   (ex overwrites logits)
__global__ void edge_exp_kernel(float* __restrict__ logits,
                                const int* __restrict__ nmax,
                                const int* __restrict__ dst,
                                float* __restrict__ denom, long long E) {
  long long t = (long long)blockIdx.x * blockDim.x + threadIdx.x;
  if (t >= E * 8) return;
  long long e = t >> 3;
  int h = (int)(t & 7);
  int d = dst[e];
  float lm = key2f(nmax[(size_t)d * 8 + h]);
  float ex = __expf(logits[t] - lm);
  logits[t] = ex;
  atomicAdd(denom + (size_t)d * 8 + h, ex);
}

// pass 3: agg[dst,c] += (ex/denom) * xl[src,c]
__global__ void edge_scatter_kernel(const float* __restrict__ ex,
                                    const float* __restrict__ denom,
                                    const float* __restrict__ xl,
                                    const int* __restrict__ src,
                                    const int* __restrict__ dst,
                                    float* __restrict__ agg, long long E) {
  long long t = (long long)blockIdx.x * blockDim.x + threadIdx.x;
  if (t >= E * 128) return;
  long long e = t >> 7;
  int c = (int)(t & 127);
  int h = c >> 4;
  int s = src[e], d = dst[e];
  float a = ex[e * 8 + h] / (denom[(size_t)d * 8 + h] + 1e-16f);
  atomicAdd(agg + (size_t)d * 128 + c, a * xl[(size_t)s * 128 + c]);
}

// ------------------------------- pooling / head ----------------------------

__global__ void pool_add_kernel(const float* __restrict__ x, const int* __restrict__ batch,
                                float* __restrict__ pooled, int colofs, size_t n) {
  size_t i = (size_t)blockIdx.x * blockDim.x + threadIdx.x;
  if (i < n) {
    size_t node = i >> 7;
    int c = (int)(i & 127);
    atomicAdd(pooled + (size_t)batch[node] * 256 + colofs + c, x[i]);
  }
}

__global__ void fc2_sigmoid_kernel(const float* __restrict__ z1, const float* __restrict__ fc2W,
                                   const float* __restrict__ fc2b, float* __restrict__ out, int B) {
  int b = blockIdx.x * blockDim.x + threadIdx.x;
  if (b >= B) return;
  float s = fc2b[0];
  for (int j = 0; j < 128; ++j) s += z1[(size_t)b * 128 + j] * fc2W[j];
  out[b] = 1.f / (1.f + __expf(-s));
}

// ------------------------------- host orchestration ------------------------

static void run_stack(const float* x_in, const float* eattr,
                      const float* linlW, const float* linlb,
                      const float* linrW, const float* linrb,
                      const float* lineW, const float* att, const float* bias,
                      const float* resW, const float* resb,
                      const int* edge_index, const int* batch,
                      int N, long long E, int ED, int colofs,
                      float* xcur, unsigned short* xbf, float* xl, float* xr,
                      float* hbuf, float* exb, int* nmax, float* denom,
                      unsigned short* Bpack, float* pooled, hipStream_t stream) {
  const int* src = edge_index;
  const int* dst = edge_index + E;
  size_t nC = (size_t)N * 128;
  unsigned gN    = (unsigned)((nC + 255) / 256);
  unsigned gNH   = (unsigned)(((size_t)N * 8 + 255) / 256);
  unsigned gE8   = (unsigned)((E * 8 + 255) / 256);
  unsigned gE128 = (unsigned)((E * 128 + 255) / 256);
  unsigned gPack = (unsigned)((4 * 8 * 32 + 255) / 256);
  unsigned tilesM = (unsigned)((N + 15) / 16);

  for (int l = 0; l < 3; ++l) {
    const float* xin = (l == 0) ? x_in : xcur;
    cvt_f32_bf16_kernel<<<gN, 256, 0, stream>>>(xin, xbf, nC);

    // xl = x @ linlW + linlb
    pack_b_kernel<<<gPack, 256, 0, stream>>>(linlW + (size_t)l * 128 * 128, Bpack, 128);
    gemm_wmma_kernel<<<tilesM, 256, 0, stream>>>(xbf, Bpack, linlb + l * 128, nullptr, xl, N, 128, 0);
    // xr = x @ linrW + linrb
    pack_b_kernel<<<gPack, 256, 0, stream>>>(linrW + (size_t)l * 128 * 128, Bpack, 128);
    gemm_wmma_kernel<<<tilesM, 256, 0, stream>>>(xbf, Bpack, linrb + l * 128, nullptr, xr, N, 128, 0);

    // init segment buffers: nmax=-inf key, denom=0, agg=0
    fill_u32_kernel<<<gNH, 256, 0, stream>>>((unsigned int*)nmax, 0x80000000u, (size_t)N * 8);
    fill_u32_kernel<<<gNH, 256, 0, stream>>>((unsigned int*)denom, 0u, (size_t)N * 8);
    fill_u32_kernel<<<gN, 256, 0, stream>>>((unsigned int*)hbuf, 0u, nC);

    edge_logits_kernel<<<gE8, 256, 0, stream>>>(xl, xr, eattr, lineW + (size_t)l * ED * 128,
                                                att + l * 128, src, dst, exb, nmax, E, ED);
    edge_exp_kernel<<<gE8, 256, 0, stream>>>(exb, nmax, dst, denom, E);
    edge_scatter_kernel<<<gE128, 256, 0, stream>>>(exb, denom, xl, src, dst, hbuf, E);

    add_bias_kernel<<<gN, 256, 0, stream>>>(hbuf, bias + l * 128, nC);

    // x = h + elu(h @ resW + resb)
    cvt_f32_bf16_kernel<<<gN, 256, 0, stream>>>(hbuf, xbf, nC);
    pack_b_kernel<<<gPack, 256, 0, stream>>>(resW + (size_t)l * 128 * 128, Bpack, 128);
    gemm_wmma_kernel<<<tilesM, 256, 0, stream>>>(xbf, Bpack, resb + l * 128, hbuf, xcur, N, 128, 1);
  }
  pool_add_kernel<<<gN, 256, 0, stream>>>(xcur, batch, pooled, colofs, nC);
}

extern "C" void kernel_launch(void* const* d_in, const int* in_sizes, int n_in,
                              void* d_out, int out_size, void* d_ws, size_t ws_size,
                              hipStream_t stream) {
  // inputs (setup_inputs order)
  const float* mg_x     = (const float*)d_in[0];
  const float* pg_x     = (const float*)d_in[1];
  const float* mg_eattr = (const float*)d_in[2];
  const float* pg_eattr = (const float*)d_in[3];
  const float* mg_linlW = (const float*)d_in[4];
  const float* mg_linlb = (const float*)d_in[5];
  const float* mg_linrW = (const float*)d_in[6];
  const float* mg_linrb = (const float*)d_in[7];
  const float* mg_lineW = (const float*)d_in[8];
  const float* mg_att   = (const float*)d_in[9];
  const float* mg_bias  = (const float*)d_in[10];
  const float* mg_resW  = (const float*)d_in[11];
  const float* mg_resb  = (const float*)d_in[12];
  const float* pg_linlW = (const float*)d_in[13];
  const float* pg_linlb = (const float*)d_in[14];
  const float* pg_linrW = (const float*)d_in[15];
  const float* pg_linrb = (const float*)d_in[16];
  const float* pg_lineW = (const float*)d_in[17];
  const float* pg_att   = (const float*)d_in[18];
  const float* pg_bias  = (const float*)d_in[19];
  const float* pg_resW  = (const float*)d_in[20];
  const float* pg_resb  = (const float*)d_in[21];
  const float* fc1W     = (const float*)d_in[22];
  const float* fc1b     = (const float*)d_in[23];
  const float* fc2W     = (const float*)d_in[24];
  const float* fc2b     = (const float*)d_in[25];
  const int* mg_ei      = (const int*)d_in[26];
  const int* pg_ei      = (const int*)d_in[27];
  const int* mg_batch   = (const int*)d_in[28];
  const int* pg_batch   = (const int*)d_in[29];

  int NM = in_sizes[0] / 128;
  int NP = in_sizes[1] / 128;
  long long EM = in_sizes[26] / 2;
  long long EP = in_sizes[27] / 2;
  int EDM = (int)(in_sizes[2] / EM);
  int EDP = (int)(in_sizes[3] / EP);
  int B = out_size;   // number of graphs (128)

  // workspace carve (256B aligned chunks)
  char* p = (char*)d_ws;
  auto carve = [&](size_t bytes) -> void* {
    void* r = (void*)p;
    p += (bytes + 255) & ~(size_t)255;
    return r;
  };
  int Nmax = NM > NP ? NM : NP;
  long long Emax = EM > EP ? EM : EP;
  float*          xcur     = (float*)carve((size_t)Nmax * 128 * 4);
  unsigned short* xbf      = (unsigned short*)carve((size_t)Nmax * 128 * 2);
  float*          xl       = (float*)carve((size_t)Nmax * 128 * 4);
  float*          xr       = (float*)carve((size_t)Nmax * 128 * 4);
  float*          hbuf     = (float*)carve((size_t)Nmax * 128 * 4);
  float*          exb      = (float*)carve((size_t)Emax * 8 * 4);
  int*            nmax     = (int*)carve((size_t)Nmax * 8 * 4);
  float*          denom    = (float*)carve((size_t)Nmax * 8 * 4);
  unsigned short* Bpack    = (unsigned short*)carve((size_t)8 * 8 * 32 * 16 * 2); // K up to 256
  float*          pooled   = (float*)carve((size_t)B * 256 * 4);
  unsigned short* pooledbf = (unsigned short*)carve((size_t)B * 256 * 2);
  float*          z1       = (float*)carve((size_t)B * 128 * 4);

  // zero pooled accumulator
  {
    size_t n = (size_t)B * 256;
    fill_u32_kernel<<<(unsigned)((n + 255) / 256), 256, 0, stream>>>((unsigned int*)pooled, 0u, n);
  }

  run_stack(mg_x, mg_eattr, mg_linlW, mg_linlb, mg_linrW, mg_linrb, mg_lineW,
            mg_att, mg_bias, mg_resW, mg_resb, mg_ei, mg_batch,
            NM, EM, EDM, /*colofs=*/0,
            xcur, xbf, xl, xr, hbuf, exb, nmax, denom, Bpack, pooled, stream);

  run_stack(pg_x, pg_eattr, pg_linlW, pg_linlb, pg_linrW, pg_linrb, pg_lineW,
            pg_att, pg_bias, pg_resW, pg_resb, pg_ei, pg_batch,
            NP, EP, EDP, /*colofs=*/128,
            xcur, xbf, xl, xr, hbuf, exb, nmax, denom, Bpack, pooled, stream);

  // z1 = elu(pooled @ fc1W + fc1b)   [B x 256] * [256 x 128]
  {
    size_t n = (size_t)B * 256;
    cvt_f32_bf16_kernel<<<(unsigned)((n + 255) / 256), 256, 0, stream>>>(pooled, pooledbf, n);
    unsigned gPack = (unsigned)((8 * 8 * 32 + 255) / 256);
    pack_b_kernel<<<gPack, 256, 0, stream>>>(fc1W, Bpack, 256);
    gemm_wmma_kernel<<<(unsigned)((B + 15) / 16), 256, 0, stream>>>(pooledbf, Bpack, fc1b,
                                                                    nullptr, z1, B, 256, 2);
  }
  // out = sigmoid(z1 @ fc2W + fc2b)
  fc2_sigmoid_kernel<<<(unsigned)((B + 127) / 128), 128, 0, stream>>>(z1, fc2W, fc2b,
                                                                      (float*)d_out, B);
}
